// Model21_82841329205453
// MI455X (gfx1250) — matrix operand order repaired and checked by reading, not compiled, not run on hardware
//
#include <hip/hip_runtime.h>
#include <hip/hip_bf16.h>
#include <math.h>

// ---------------------------------------------------------------- types
typedef __bf16 bf16_t;
typedef bf16_t v16bf __attribute__((ext_vector_type(16)));
typedef bf16_t v8bf  __attribute__((ext_vector_type(8)));
typedef float  v8f   __attribute__((ext_vector_type(8)));
typedef unsigned int u32x4 __attribute__((ext_vector_type(4)));
typedef int          i32x4 __attribute__((ext_vector_type(4)));
typedef int          i32x8 __attribute__((ext_vector_type(8)));

__device__ __forceinline__ v8f zero8() {
    v8f z;
#pragma unroll
    for (int i = 0; i < 8; ++i) z[i] = 0.0f;
    return z;
}

__device__ __forceinline__ float gelu_exact(float x) {
    return 0.5f * x * (1.0f + erff(x * 0.70710678118654752440f));
}

// WMMA: D = A(16x32 bf16) * B(32x16 bf16) + C(16x16 f32)
__device__ __forceinline__ v8f wmma_bf16(v16bf a, v16bf b, v8f c) {
    return __builtin_amdgcn_wmma_f32_16x16x32_bf16(
        false, a, false, b, (short)0, c, false, false);
}

// A-frag (16x32, MxK) from row-major [row][k] memory (ISA 7.12.2):
// lane L: row = L%16, chunk0 = k[8h..8h+7], chunk1 = k[16+8h..16+8h+7]
__device__ __forceinline__ v16bf load_a_frag(const bf16_t* base, int rs, int lane) {
    const int row = lane & 15, half = lane >> 4;
    const bf16_t* p = base + row * rs + 8 * half;
    v8bf c0 = *(const v8bf*)(p);
    v8bf c1 = *(const v8bf*)(p + 16);
    return __builtin_shufflevector(c0, c1, 0,1,2,3,4,5,6,7,8,9,10,11,12,13,14,15);
}

// B-frag (32x16, KxN) from memory laid out [col][k] row-major:
// lane L: col = L%16, chunk0 = k[8h..8h+7], chunk1 = k[16+8h..16+8h+7]
__device__ __forceinline__ v16bf load_b_frag(const bf16_t* base, int rs, int lane) {
    const int col = lane & 15, half = lane >> 4;
    const bf16_t* p = base + col * rs + 8 * half;
    v8bf c0 = *(const v8bf*)(p);
    v8bf c1 = *(const v8bf*)(p + 16);
    return __builtin_shufflevector(c0, c1, 0,1,2,3,4,5,6,7,8,9,10,11,12,13,14,15);
}

// ---- CDNA5 async global->LDS b128 copy (ASYNCcnt-tracked, no VGPR data path)
__device__ __forceinline__ void async_b128(void* lds, const void* g) {
    unsigned l = (unsigned)(unsigned long long)lds;  // LDS byte address = addr[31:0]
    asm volatile("global_load_async_to_lds_b128 %0, %1, off"
                 :: "v"(l), "v"((unsigned long long)g) : "memory");
}
__device__ __forceinline__ void wait_async0() {
    asm volatile("s_wait_asynccnt 0x0" ::: "memory");
}

// ---- CDNA5 Tensor Data Mover: 2D bf16 tile -> LDS with row padding (ISA ch8)
// Descriptor per D# groups 0/1; groups 2/3 zero (tensor <= 2D).
__device__ __forceinline__ void tdm_load_2d(const void* gptr, void* lds,
        unsigned tile_d0, unsigned tile_d1,
        unsigned tensor_d0, unsigned tensor_d1,
        unsigned stride0_elems,
        unsigned pad_interval_enc, unsigned pad_amount_enc)
{
    const unsigned long long ga = (unsigned long long)gptr;
    const unsigned la = (unsigned)(unsigned long long)lds;

    u32x4 g0;
    g0[0] = 1u;                                             // count=1 (valid D#)
    g0[1] = la;                                             // lds_addr
    g0[2] = (unsigned)(ga & 0xFFFFFFFFu);                   // global_addr[31:0]
    g0[3] = (unsigned)((ga >> 32) & 0x01FFFFFFu) | (2u << 30); // addr[56:32] | type=2

    i32x8 g1;
    g1[0] = (int)((1u << 16)                                // data_size = 2 bytes
                | (1u << 20)                                // pad_enable
                | (pad_interval_enc << 22)
                | (pad_amount_enc << 25));
    g1[1] = (int)((tensor_d0 & 0xFFFFu) << 16);             // tensor_dim0[15:0]
    g1[2] = (int)((tensor_d0 >> 16) | ((tensor_d1 & 0xFFFFu) << 16));
    g1[3] = (int)((tensor_d1 >> 16) | ((tile_d0 & 0xFFFFu) << 16)); // tile_dim0
    g1[4] = (int)(tile_d1 & 0xFFFFu);                       // tile_dim1 (tile_dim2=0)
    g1[5] = (int)stride0_elems;                             // tensor_dim0_stride[31:0]
    g1[6] = 0;                                              // stride0 hi / stride1 lo
    g1[7] = 0;

    i32x4 gz;  gz[0] = gz[1] = gz[2] = gz[3] = 0;
#if defined(__clang_major__) && (__clang_major__ >= 23)
    i32x8 gz8;
#pragma unroll
    for (int i = 0; i < 8; ++i) gz8[i] = 0;
    __builtin_amdgcn_tensor_load_to_lds(g0, g1, gz, gz, gz8, 0);
#else
    __builtin_amdgcn_tensor_load_to_lds(g0, g1, gz, gz, 0);
#endif
}
__device__ __forceinline__ void wait_tensor0() {
    __builtin_amdgcn_s_wait_tensorcnt((short)0);
}

// ---- DPP16 butterfly reductions over 16-lane halves (basis {1,2,7,15})
#define DPP_STEP(x, ctrl) __builtin_amdgcn_update_dpp(0, (x), (ctrl), 0xF, 0xF, true)
__device__ __forceinline__ float dpp_max16(float x) {
    union { float f; int i; } a, b;
    a.f = x;
    b.i = DPP_STEP(a.i, 0xB1);  a.f = fmaxf(a.f, b.f);  // xor 1 (quad_perm 1,0,3,2)
    b.i = DPP_STEP(a.i, 0x4E);  a.f = fmaxf(a.f, b.f);  // xor 2 (quad_perm 2,3,0,1)
    b.i = DPP_STEP(a.i, 0x141); a.f = fmaxf(a.f, b.f);  // xor 7 (row_half_mirror)
    b.i = DPP_STEP(a.i, 0x140); a.f = fmaxf(a.f, b.f);  // xor 15 (row_mirror)
    return a.f;
}
__device__ __forceinline__ float dpp_sum16(float x) {
    union { float f; int i; } a, b;
    a.f = x;
    b.i = DPP_STEP(a.i, 0xB1);  a.f += b.f;
    b.i = DPP_STEP(a.i, 0x4E);  a.f += b.f;
    b.i = DPP_STEP(a.i, 0x141); a.f += b.f;
    b.i = DPP_STEP(a.i, 0x140); a.f += b.f;
    return a.f;
}

// ---------------------------------------------------------------- sizes
constexpr size_t NB = 4, DIM = 512, NSEQ = 2048, HEADS = 8, DH = 64, QKVC = 1536;
constexpr size_t MTOT = NB * NSEQ;  // 8192 rows in all GEMMs

constexpr size_t XT_OFF   = 0;
constexpr size_t WQKV_OFF = XT_OFF   + NB*NSEQ*DIM*2;
constexpr size_t WOUT_OFF = WQKV_OFF + QKVC*DIM*2;
constexpr size_t WFFN_OFF = WOUT_OFF + DIM*DIM*2;
constexpr size_t Q_OFF    = WFFN_OFF + DIM*DIM*2;
constexpr size_t K_OFF    = Q_OFF    + NB*HEADS*NSEQ*DH*2;
constexpr size_t V_OFF    = K_OFF    + NB*HEADS*NSEQ*DH*2;   // V stored [bh][dh][n]
constexpr size_t O_OFF    = V_OFF    + NB*HEADS*NSEQ*DH*2;
constexpr size_t ATTF_OFF = O_OFF    + NB*NSEQ*DIM*2;
constexpr size_t ATTB_OFF = ATTF_OFF + NB*NSEQ*DIM*4;

// ---------------------------------------------------------------- prep
__global__ void convert_f32_to_bf16(const float* __restrict__ src,
                                    bf16_t* __restrict__ dst, int total) {
    int idx = blockIdx.x * blockDim.x + threadIdx.x;
    if (idx < total) dst[idx] = (bf16_t)src[idx];
}
// dst[c*R + r] = (bf16) src[r*C + c]
__global__ void transpose_f32_to_bf16(const float* __restrict__ src,
                                      bf16_t* __restrict__ dst, int R, int C) {
    int idx = blockIdx.x * blockDim.x + threadIdx.x;
    if (idx >= R * C) return;
    int r = idx / C, c = idx - r * C;
    dst[(size_t)c * R + r] = (bf16_t)src[idx];
}

// ---------------------------------------------------------------- GEMM
// C[8192, Ncols] = A[8192,512] * W[Ncols,512]^T.  Both 128x32 tiles are moved
// by the Tensor Data Mover (one tensor_load_to_lds per matrix per K-step,
// issued by wave 0, TENSORcnt-tracked, LDS row-padded to stride 40 by TDM).
#define BM 128
#define BN 128
#define BK 32

template <int MODE>
__global__ __launch_bounds__(256) void gemm_bf16_kernel(
    const bf16_t* __restrict__ A, const bf16_t* __restrict__ Bm, int Ncols,
    bf16_t* __restrict__ qs, bf16_t* __restrict__ kb, bf16_t* __restrict__ vb,
    const float* __restrict__ bias, const float* __restrict__ g,
    const float* __restrict__ be,   const float* __restrict__ mu,
    const float* __restrict__ var,
    const float* __restrict__ xin, float* __restrict__ attF, bf16_t* __restrict__ attB,
    const float* __restrict__ attFin, float* __restrict__ outp)
{
    __shared__ bf16_t ldsA [2][BM][BK + 8];   // 80B rows (TDM pad 4DW / 16DW)
    __shared__ bf16_t ldsBt[2][BN][BK + 8];   // [col][k]

    const int tid  = threadIdx.x;
    const int lane = tid & 31;
    const int wid  = tid >> 5;
    const int mBase = (wid & 3) * 32;
    const int nBase = (wid >> 2) * 64;
    const int rowBlock = blockIdx.x * BM;
    const int colBlock = blockIdx.y * BN;
    const int Kdim = 512;

    // TDM stage: 128x32 bf16 tiles of A and W; pad 16DW interval(enc 3), 4DW amount(enc 3)
    auto stage = [&](int k0, int buf) {
        tdm_load_2d(A  + (size_t)rowBlock * Kdim + k0, &ldsA [buf][0][0],
                    BK, BM, Kdim, BM, Kdim, 3, 3);
        tdm_load_2d(Bm + (size_t)colBlock * Kdim + k0, &ldsBt[buf][0][0],
                    BK, BN, Kdim, BN, Kdim, 3, 3);
    };

    v8f acc[2][4];
#pragma unroll
    for (int i = 0; i < 2; ++i)
#pragma unroll
        for (int j = 0; j < 4; ++j) acc[i][j] = zero8();

    if (wid == 0) stage(0, 0);
    wait_tensor0();
    __syncthreads();

    int buf = 0;
    for (int k0 = 0; k0 < Kdim; k0 += BK) {
        if (k0 + BK < Kdim && wid == 0) stage(k0 + BK, buf ^ 1);  // overlap DMA

        v16bf af[2], bfr[4];
#pragma unroll
        for (int tm = 0; tm < 2; ++tm)
            af[tm] = load_a_frag(&ldsA[buf][mBase + tm * 16][0], BK + 8, lane);
#pragma unroll
        for (int tn = 0; tn < 4; ++tn)
            bfr[tn] = load_b_frag(&ldsBt[buf][nBase + tn * 16][0], BK + 8, lane);
#pragma unroll
        for (int tm = 0; tm < 2; ++tm)
#pragma unroll
            for (int tn = 0; tn < 4; ++tn)
                acc[tm][tn] = wmma_bf16(af[tm], bfr[tn], acc[tm][tn]);

        wait_tensor0();          // no-op for waves 1..7 (TENSORcnt==0)
        __syncthreads();
        buf ^= 1;
    }

    // ------------- epilogue (C layout: row = v + 8*half, col = lane%16)
    const int half = lane >> 4;
#pragma unroll
    for (int tm = 0; tm < 2; ++tm)
#pragma unroll
        for (int tn = 0; tn < 4; ++tn)
#pragma unroll
            for (int v = 0; v < 8; ++v) {
                const int R  = rowBlock + mBase + tm * 16 + v + 8 * half;
                const int Cc = colBlock + nBase + tn * 16 + (lane & 15);
                const int b  = R >> 11, nn = R & 2047;
                const float val = acc[tm][tn][v];
                if (MODE == 0) {
                    const int o = Cc;
                    if (o < 512) {           // Q [bh][n][dh], pre-scaled by dh^-0.5
                        qs[(((size_t)b * 8 + (o >> 6)) * NSEQ + nn) * DH + (o & 63)] =
                            (bf16_t)(val * 0.125f);
                    } else if (o < 1024) {   // K [bh][n][dh]
                        const int oo = o - 512;
                        kb[(((size_t)b * 8 + (oo >> 6)) * NSEQ + nn) * DH + (oo & 63)] = (bf16_t)val;
                    } else {                 // V stored TRANSPOSED: [bh][dh][n]
                        const int oo = o - 1024;
                        vb[(((size_t)b * 8 + (oo >> 6)) * DH + (oo & 63)) * NSEQ + nn] = (bf16_t)val;
                    }
                } else if (MODE == 1) {
                    const int o = Cc;
                    const float inv = g[o] * rsqrtf(var[o] + 1e-5f);
                    float t = (val + bias[o]) * inv + (be[o] - mu[o] * inv);
                    t += xin[((size_t)b * DIM + o) * NSEQ + nn];
                    const float gl = gelu_exact(t);
                    const size_t oi = ((size_t)b * NSEQ + nn) * DIM + o;
                    attF[oi] = gl;
                    attB[oi] = (bf16_t)gl;
                } else {
                    const int o = Cc;
                    const float inv = g[o] * rsqrtf(var[o] + 1e-5f);
                    const float f = val * inv + (be[o] - mu[o] * inv);
                    const float a = attFin[((size_t)b * NSEQ + nn) * DIM + o];
                    outp[((size_t)b * DIM + o) * NSEQ + nn] = gelu_exact(a + f);
                }
            }
}

// ---------------------------------------------------------------- attention
// Flash attention: grid (16 qblocks, 32 bh), 8 waves/block, 16 q-rows/wave.
// K [bh][n][dh] and V^T [bh][dh][n] tiles double-buffered via async b128 DMA.
__global__ __launch_bounds__(256) void attention_kernel(
    const bf16_t* __restrict__ Qs, const bf16_t* __restrict__ Kb,
    const bf16_t* __restrict__ Vb, bf16_t* __restrict__ Ob)
{
    __shared__ bf16_t ldsK [2][32][72];     // K tile [key][dh]
    __shared__ bf16_t ldsVt[2][64][40];     // V tile [dh][key]
    __shared__ bf16_t ldsP [8][16][40];     // per-wave P (16x32) bounce buffer

    const int tid = threadIdx.x, lane = tid & 31, wid = tid >> 5;
    const int half = lane >> 4;
    const int bh = blockIdx.y, b = bh >> 3, h = bh & 7;
    const bf16_t* Qp = Qs + (size_t)bh * NSEQ * DH;
    const bf16_t* Kp = Kb + (size_t)bh * NSEQ * DH;
    const bf16_t* Vp = Vb + (size_t)bh * DH * NSEQ;   // [dh][n]
    const int q0 = blockIdx.x * 128 + wid * 16;

    // stage one 32-key tile: 256 + 256 b128 chunks -> 1 K + 1 V chunk per thread
    const int krow = tid >> 3, kko = (tid & 7) * 8;   // K: [32][64]
    const int vrow = tid >> 2, vko = (tid & 3) * 8;   // Vt: [64][32]
    auto stage = [&](int j0, int buf) {
        async_b128(&ldsK [buf][krow][kko], Kp + (size_t)(j0 + krow) * DH + kko);
        async_b128(&ldsVt[buf][vrow][vko], Vp + (size_t)vrow * NSEQ + j0 + vko);
    };

    // Q fragments for this wave's 16 rows (dh 0..31 / 32..63), in registers
    const v16bf qf0 = load_a_frag(Qp + (size_t)q0 * DH,      DH, lane);
    const v16bf qf1 = load_a_frag(Qp + (size_t)q0 * DH + 32, DH, lane);

    v8f oacc[4];
#pragma unroll
    for (int c = 0; c < 4; ++c) oacc[c] = zero8();
    float mrow[8], lrow[8];
#pragma unroll
    for (int v = 0; v < 8; ++v) { mrow[v] = -__builtin_inff(); lrow[v] = 0.0f; }

    stage(0, 0);
    wait_async0();
    __syncthreads();

    int buf = 0;
    for (int j0 = 0; j0 < (int)NSEQ; j0 += 32) {
        if (j0 + 32 < (int)NSEQ) stage(j0 + 32, buf ^ 1);  // overlap DMA

        // S = Q * K^T for 2 x 16-key sub-tiles
        v8f s[2];
#pragma unroll
        for (int jt = 0; jt < 2; ++jt) {
            s[jt] = zero8();
            s[jt] = wmma_bf16(qf0, load_b_frag(&ldsK[buf][jt * 16][0],  72, lane), s[jt]);
            s[jt] = wmma_bf16(qf1, load_b_frag(&ldsK[buf][jt * 16][32], 72, lane), s[jt]);
        }

        // online softmax (DPP16 butterflies, no LDS traffic)
        float al[8];
#pragma unroll
        for (int v = 0; v < 8; ++v) {
            const float t  = dpp_max16(fmaxf(s[0][v], s[1][v]));
            const float mn = fmaxf(mrow[v], t);
            al[v] = __expf(mrow[v] - mn);
            const float p0 = __expf(s[0][v] - mn);
            const float p1 = __expf(s[1][v] - mn);
            lrow[v] = lrow[v] * al[v] + dpp_sum16(p0 + p1);
            mrow[v] = mn;
            ldsP[wid][v + 8 * half][lane & 15]        = (bf16_t)p0;
            ldsP[wid][v + 8 * half][16 + (lane & 15)] = (bf16_t)p1;
        }
#pragma unroll
        for (int c = 0; c < 4; ++c)
#pragma unroll
            for (int v = 0; v < 8; ++v) oacc[c][v] = oacc[c][v] * al[v];

        asm volatile("s_wait_dscnt 0x0" ::: "memory");  // same-wave P visibility
        const v16bf pf = load_a_frag(&ldsP[wid][0][0], 40, lane);
#pragma unroll
        for (int c = 0; c < 4; ++c)
            oacc[c] = wmma_bf16(pf, load_b_frag(&ldsVt[buf][c * 16][0], 40, lane), oacc[c]);

        wait_async0();
        __syncthreads();
        buf ^= 1;
    }

    // O /= l, write to [b, n, inner]
#pragma unroll
    for (int c = 0; c < 4; ++c)
#pragma unroll
        for (int v = 0; v < 8; ++v) {
            const int row = q0 + v + 8 * half;
            const int col = h * 64 + c * 16 + (lane & 15);
            Ob[((size_t)b * NSEQ + row) * DIM + col] = (bf16_t)(oacc[c][v] / lrow[v]);
        }
}

// ---------------------------------------------------------------- launch
extern "C" void kernel_launch(void* const* d_in, const int* in_sizes, int n_in,
                              void* d_out, int out_size, void* d_ws, size_t ws_size,
                              hipStream_t stream) {
    (void)in_sizes; (void)n_in; (void)out_size; (void)ws_size;
    const float* x     = (const float*)d_in[0];
    const float* W_qkv = (const float*)d_in[1];
    const float* W_out = (const float*)d_in[2];
    const float* b_out = (const float*)d_in[3];
    const float* g1    = (const float*)d_in[4];
    const float* beta1 = (const float*)d_in[5];
    const float* m1    = (const float*)d_in[6];
    const float* v1    = (const float*)d_in[7];
    const float* W_ffn = (const float*)d_in[8];
    const float* g2    = (const float*)d_in[9];
    const float* beta2 = (const float*)d_in[10];
    const float* m2    = (const float*)d_in[11];
    const float* v2    = (const float*)d_in[12];
    float* out = (float*)d_out;

    char* ws = (char*)d_ws;
    bf16_t* xt    = (bf16_t*)(ws + XT_OFF);
    bf16_t* wqkvB = (bf16_t*)(ws + WQKV_OFF);   // [1536][512], original layout
    bf16_t* woutB = (bf16_t*)(ws + WOUT_OFF);   // [512][512]
    bf16_t* wffnB = (bf16_t*)(ws + WFFN_OFF);
    bf16_t* qs    = (bf16_t*)(ws + Q_OFF);
    bf16_t* kb    = (bf16_t*)(ws + K_OFF);
    bf16_t* vb    = (bf16_t*)(ws + V_OFF);
    bf16_t* ob    = (bf16_t*)(ws + O_OFF);
    float*  attF  = (float*) (ws + ATTF_OFF);
    bf16_t* attB  = (bf16_t*)(ws + ATTB_OFF);

    const dim3 blk(256);
    // weights: straight bf16 convert, keep [o][k] layout (== B-frag orientation)
    convert_f32_to_bf16<<<(QKVC*DIM + 255) / 256, blk, 0, stream>>>(W_qkv, wqkvB, QKVC*DIM);
    convert_f32_to_bf16<<<(DIM*DIM  + 255) / 256, blk, 0, stream>>>(W_out, woutB, DIM*DIM);
    convert_f32_to_bf16<<<(DIM*DIM  + 255) / 256, blk, 0, stream>>>(W_ffn, wffnB, DIM*DIM);
    // x[b][d][n] -> xt[b][n][d] bf16
    for (int b = 0; b < (int)NB; ++b)
        transpose_f32_to_bf16<<<(DIM*NSEQ + 255) / 256, blk, 0, stream>>>(
            x + (size_t)b * DIM * NSEQ, xt + (size_t)b * NSEQ * DIM, DIM, NSEQ);

    // QKV projection (+ head scatter, Q scale, V transposed)
    gemm_bf16_kernel<0><<<dim3(MTOT / BM, QKVC / BN), blk, 0, stream>>>(
        xt, wqkvB, (int)QKVC, qs, kb, vb,
        nullptr, nullptr, nullptr, nullptr, nullptr,
        nullptr, nullptr, nullptr, nullptr, nullptr);

    // flash attention
    attention_kernel<<<dim3(NSEQ / 128, NB * HEADS), blk, 0, stream>>>(qs, kb, vb, ob);

    // out-proj + bias + BN1 + residual + GELU
    gemm_bf16_kernel<1><<<dim3(MTOT / BM, DIM / BN), blk, 0, stream>>>(
        ob, woutB, (int)DIM, nullptr, nullptr, nullptr,
        b_out, g1, beta1, m1, v1,
        x, attF, attB, nullptr, nullptr);

    // FFN + BN2 + residual + GELU -> d_out [b, d, n]
    gemm_bf16_kernel<2><<<dim3(MTOT / BM, DIM / BN), blk, 0, stream>>>(
        attB, wffnB, (int)DIM, nullptr, nullptr, nullptr,
        nullptr, g2, beta2, m2, v2,
        nullptr, nullptr, nullptr, attF, out);
}
